// Mamba2HSIClassifier_11871289606256
// MI455X (gfx1250) — matrix operand order repaired and checked by reading, not compiled, not run on hardware
//
#include <hip/hip_runtime.h>
#include <hip/hip_bf16.h>
#include <math.h>

// ---------------------------------------------------------------------------
// Mamba2 HSI classifier forward on gfx1250 (MI455X).
// - all matmuls on v_wmma_f32_16x16x32_f16 with staged/pre-swizzled f16
//   operands (contiguous b128 loads, zero branches in the hot loops)
// - SSD output kernel stages its shared 64x256 x-tile into LDS with
//   global_load_async_to_lds_b128 (+ s_wait_asynccnt), reads it via DS
// ---------------------------------------------------------------------------

typedef __attribute__((ext_vector_type(16))) _Float16 v16h;
typedef __attribute__((ext_vector_type(8)))  _Float16 v8h;
typedef __attribute__((ext_vector_type(8)))  float    v8f;

#define HIMG 145
#define WIMG 145
#define LSEQ (145*145)          // 21025
#define LPAD 21248              // 83 * 256
#define NCK  83
#define DMODEL 256
#define DINNER 512
#define DSTATE 128
#define NHEADS 8
#define HEADDIM 64
#define DCONV 4
#define CONVDIM 768             // DINNER + 2*DSTATE
#define DINPROJ 1288            // 2*DINNER + 2*DSTATE + NHEADS
#define BANDS 200
#define NCLS 17
#define KCONV 1800              // 9*BANDS
#define KCONVP 1824             // padded to 32
#define MPAD 21040              // LSEQ rounded up to 16

__device__ __forceinline__ v8f wmma32(v16h a, v16h b, v8f c) {
  return __builtin_amdgcn_wmma_f32_16x16x32_f16(false, a, false, b, (short)0, c, false, false);
}
__device__ __forceinline__ float geluf(float x) {
  return 0.5f * x * (1.0f + erff(x * 0.70710678118654752f));
}
__device__ __forceinline__ float siluf(float x) {
  return x / (1.0f + __expf(-x));
}
__device__ __forceinline__ float softplusf(float x) {
  return (x > 20.0f) ? x : log1pf(__expf(x));
}
__device__ __forceinline__ v16h joinh(v8h lo, v8h hi) {
  return __builtin_shufflevector(lo, hi, 0, 1, 2, 3, 4, 5, 6, 7,
                                 8, 9, 10, 11, 12, 13, 14, 15);
}

// ------------------- f32 -> padded f16 activation staging ------------------
__global__ __launch_bounds__(256)
void k_cvtA(const float* __restrict__ A, _Float16* __restrict__ A16,
            int M, int K, int Kp, int tot)
{
  int idx = blockIdx.x * blockDim.x + threadIdx.x;
  if (idx >= tot) return;
  int row = idx / Kp, col = idx - row * Kp;
  float v = (row < M && col < K) ? A[(size_t)row * K + col] : 0.0f;
  A16[idx] = (_Float16)v;
}

// ------------------- weight pre-swizzle into B-fragment layout -------------
__global__ __launch_bounds__(256)
void k_packW(const float* __restrict__ W, _Float16* __restrict__ Wp,
             int K, int N, int nkb, int tot)
{
  int idx = blockIdx.x * blockDim.x + threadIdx.x;
  if (idx >= tot) return;
  int j = idx & 15;
  int lane = (idx >> 4) & 31;
  int rem = idx >> 9;
  int kb = rem % nkb;
  int tn = rem / nkb;
  int n = tn * 16 + (lane & 15);
  int k = kb * 32 + (lane >> 4) * 16 + j;
  float v = (k < K && n < N) ? W[(size_t)k * N + n] : 0.0f;
  Wp[idx] = (_Float16)v;
}

// ------------------- clean WMMA GEMM on staged f16 operands ----------------
__global__ __launch_bounds__(256)
void k_gemm16(const _Float16* __restrict__ A16, const _Float16* __restrict__ Wp,
              const float* __restrict__ bias, const float* __restrict__ scale,
              const float* __restrict__ shift, float smul,
              float* __restrict__ C, int M, int N, int Kp, int act)
{
  int wave = (int)((blockIdx.x * blockDim.x + threadIdx.x) >> 5);
  int lane = threadIdx.x & 31;
  int ntile = (N + 15) >> 4;
  int mtile = (M + 15) >> 4;
  if (wave >= ntile * mtile) return;              // wave-uniform
  int tm = wave / ntile, tn = wave % ntile;
  int lh = lane & 15, hg = lane >> 4;
  int nkb = Kp >> 5;
  const _Float16* Arow = A16 + (size_t)(tm * 16 + lh) * Kp + hg * 16;
  const _Float16* Wt = Wp + ((size_t)tn * nkb * 32 + lane) * 16;
  v8f acc = {};
  for (int kb = 0; kb < nkb; kb++) {
    __builtin_prefetch((const void*)(Arow + kb * 32 + 512), 0, 0);
    v8h alo = *(const v8h*)(Arow + kb * 32);
    v8h ahi = *(const v8h*)(Arow + kb * 32 + 8);
    v16h a = joinh(alo, ahi);
    v16h b = *(const v16h*)(Wt + (size_t)kb * 512);
    acc = wmma32(a, b, acc);
  }
  int col = tn * 16 + lh;
  if (col < N) {
    float bs = bias  ? bias[col]        : 0.0f;
    float sc = scale ? scale[col] * smul : 1.0f;
    float sh = shift ? shift[col]       : 0.0f;
#pragma unroll
    for (int r = 0; r < 8; r++) {
      int orow = tm * 16 + hg * 8 + r;
      if (orow < M) {
        float v = (acc[r] + bs) * sc + sh;
        if (act == 1) v = geluf(v);
        else if (act == 2) v = siluf(v);
        C[(size_t)orow * N + col] = v;
      }
    }
  }
}

// ------------------- im2col for 3x3 SAME conv (f16, padded) ----------------
__global__ __launch_bounds__(256)
void k_im2col(const float* __restrict__ xn, _Float16* __restrict__ A16)
{
  int idx = blockIdx.x * blockDim.x + threadIdx.x;
  if (idx >= MPAD * KCONVP) return;
  int row = idx / KCONVP, col = idx - row * KCONVP;
  float v = 0.0f;
  if (row < LSEQ && col < KCONV) {
    int tap = col / BANDS;
    int band = col - tap * BANDS;
    int y = row / WIMG + tap / 3 - 1;
    int x = row % WIMG + tap % 3 - 1;
    if (y >= 0 && y < HIMG && x >= 0 && x < WIMG)
      v = xn[((size_t)y * WIMG + x) * BANDS + band];
  }
  A16[idx] = (_Float16)v;
}

// ------------------------- layernorm (optionally + gelu) -------------------
__global__ __launch_bounds__(256)
void k_layernorm(const float* __restrict__ X, const float* __restrict__ w,
                 const float* __restrict__ b, float* __restrict__ Y,
                 int M, int D, int do_gelu)
{
  int wid = (int)((blockIdx.x * blockDim.x + threadIdx.x) >> 5);
  int lane = threadIdx.x & 31;
  if (wid >= M) return;
  const float* xr = X + (size_t)wid * D;
  float s = 0.0f, ss = 0.0f;
  for (int d = lane; d < D; d += 32) { float v = xr[d]; s += v; ss += v * v; }
  for (int o = 16; o > 0; o >>= 1) { s += __shfl_xor(s, o); ss += __shfl_xor(ss, o); }
  float m = s / (float)D;
  float var = ss / (float)D - m * m;
  float inv = rsqrtf(var + 1e-5f);
  float* yr = Y + (size_t)wid * D;
  for (int d = lane; d < D; d += 32) {
    float v = (xr[d] - m) * inv * w[d] + b[d];
    if (do_gelu) v = geluf(v);
    yr[d] = v;
  }
}

// ------------------------- mamba: depthwise causal conv1d + silu -----------
__global__ __launch_bounds__(256)
void k_conv1d(const float* __restrict__ ZX, const float* __restrict__ cw,
              const float* __restrict__ cb, float* __restrict__ out)
{
  int idx = blockIdx.x * blockDim.x + threadIdx.x;
  if (idx >= LSEQ * CONVDIM) return;
  int l = idx / CONVDIM, c = idx % CONVDIM;
  float acc = cb[c];
#pragma unroll
  for (int k = 0; k < DCONV; k++) {
    int ls = l - 3 + k;
    if (ls >= 0) acc += ZX[(size_t)ls * DINPROJ + DINNER + c] * cw[c * DCONV + k];
  }
  out[idx] = siluf(acc);
}

// ------------------------- mamba: B16/C16 row-major + dt -------------------
__global__ __launch_bounds__(256)
void k_pack(const float* __restrict__ xbc, const float* __restrict__ ZX,
            const float* __restrict__ dt_bias,
            _Float16* __restrict__ B16, _Float16* __restrict__ C16,
            float* __restrict__ dtp)
{
  int lp = blockIdx.x;              // [0, LPAD)
  int t = threadIdx.x;
  bool v = (lp < LSEQ);
  for (int d = t; d < DSTATE; d += 256) {
    B16[(size_t)lp * DSTATE + d] =
        (_Float16)(v ? xbc[(size_t)lp * CONVDIM + DINNER + d] : 0.0f);
    C16[(size_t)lp * DSTATE + d] =
        (_Float16)(v ? xbc[(size_t)lp * CONVDIM + DINNER + DSTATE + d] : 0.0f);
  }
  if (t < NHEADS)
    dtp[(size_t)lp * NHEADS + t] =
        v ? softplusf(ZX[(size_t)lp * DINPROJ + (DINPROJ - NHEADS) + t] + dt_bias[t]) : 0.0f;
}

// ------------------------- transposed x tile: xhT[c][h][p][q] --------------
__global__ __launch_bounds__(256)
void k_xT(const float* __restrict__ xbc, _Float16* __restrict__ xhT16)
{
  int idx = blockIdx.x * blockDim.x + threadIdx.x;
  if (idx >= LPAD * DINNER) return;
  int q = idx & 255;
  int p = (idx >> 8) & 63;
  int h = (idx >> 14) & 7;
  int c = idx >> 17;
  int lp = c * 256 + q;
  int d = h * HEADDIM + p;
  float v = (lp < LSEQ) ? xbc[(size_t)lp * CONVDIM + d] : 0.0f;
  xhT16[idx] = (_Float16)v;
}

// ------------------------- transposed B tile: BT[c][n][q] ------------------
__global__ __launch_bounds__(256)
void k_bT(const float* __restrict__ xbc, _Float16* __restrict__ BT16)
{
  int idx = blockIdx.x * blockDim.x + threadIdx.x;
  if (idx >= LPAD * DSTATE) return;
  int q = idx & 255;
  int n = (idx >> 8) & 127;
  int c = idx >> 15;
  int lp = c * 256 + q;
  float v = (lp < LSEQ) ? xbc[(size_t)lp * CONVDIM + DINNER + n] : 0.0f;
  BT16[idx] = (_Float16)v;
}

// ---------------- SSD: cumsum of dt*A, transposed dt/dacs ------------------
__global__ __launch_bounds__(64)
void k_cumsum(const float* __restrict__ dtp, const float* __restrict__ A_log,
              float* __restrict__ dacsT, float* __restrict__ dtpT,
              float* __restrict__ decay)
{
  int idx = blockIdx.x * blockDim.x + threadIdx.x;
  if (idx >= NCK * NHEADS) return;
  int c = idx >> 3, h = idx & 7;
  float Ah = -__expf(A_log[h]);
  float run = 0.0f;
  for (int q = 0; q < 256; q++) {
    float dv = dtp[((size_t)c * 256 + q) * NHEADS + h];
    run += dv * Ah;
    dacsT[(size_t)idx * 256 + q] = run;
    dtpT[(size_t)idx * 256 + q] = dv;
  }
  decay[idx] = __expf(run);
}

// -------------- SSD: xwT = xhT * dt * exp(dacs_last - dacs) ----------------
__global__ __launch_bounds__(256)
void k_xwT(const _Float16* __restrict__ xhT16, const float* __restrict__ dtpT,
           const float* __restrict__ dacsT, _Float16* __restrict__ xwT16)
{
  int idx = blockIdx.x * blockDim.x + threadIdx.x;
  if (idx >= LPAD * DINNER) return;
  int q = idx & 255;
  int ch = idx >> 14;                       // c*8 + h
  float dlast = dacsT[(size_t)ch * 256 + 255];
  float wd = dtpT[(size_t)ch * 256 + q] * __expf(dlast - dacsT[(size_t)ch * 256 + q]);
  xwT16[idx] = (_Float16)((float)xhT16[idx] * wd);
}

// ------------- SSD: chunk states (WMMA 64x128, K=256, all b128) ------------
__global__ __launch_bounds__(256)
void k_states(const _Float16* __restrict__ xwT16, const _Float16* __restrict__ BT16,
              float* __restrict__ states)
{
  int blk = blockIdx.x;                 // c*8 + h
  int c = blk >> 3, h = blk & 7;
  int wave = threadIdx.x >> 5, lane = threadIdx.x & 31;
  int lh = lane & 15, hg = lane >> 4;
  for (int t = wave; t < 32; t += 8) {  // 4 p-tiles x 8 n-tiles
    int tp = t >> 3, tn = t & 7;
    int prow = tp * 16 + lh;
    int ncol = tn * 16 + lh;
    const _Float16* Arow = xwT16 + ((size_t)blk * HEADDIM + prow) * 256 + hg * 16;
    const _Float16* Brow = BT16 + ((size_t)c * DSTATE + ncol) * 256 + hg * 16;
    v8f acc = {};
    for (int q0 = 0; q0 < 256; q0 += 32) {
      v8h alo = *(const v8h*)(Arow + q0);
      v8h ahi = *(const v8h*)(Arow + q0 + 8);
      v16h a = joinh(alo, ahi);
      v16h b = *(const v16h*)(Brow + q0);
      acc = wmma32(a, b, acc);
    }
#pragma unroll
    for (int r = 0; r < 8; r++) {
      int p = tp * 16 + hg * 8 + r;
      states[((size_t)blk * HEADDIM + p) * DSTATE + ncol] = acc[r];
    }
  }
}

// ------------------------- SSD: inter-chunk state scan ---------------------
__global__ __launch_bounds__(256)
void k_recur(const float* __restrict__ states, const float* __restrict__ decay,
             _Float16* __restrict__ prev16)
{
  int idx = blockIdx.x * blockDim.x + threadIdx.x;   // h*8192 + p*128 + n
  if (idx >= NHEADS * HEADDIM * DSTATE) return;
  int h = idx >> 13;
  float hv = 0.0f;
  for (int c = 0; c < NCK; c++) {
    size_t o = (size_t)c * (NHEADS * HEADDIM * DSTATE) + idx;
    prev16[o] = (_Float16)hv;
    hv = hv * decay[c * NHEADS + h] + states[o];
  }
}

// ------------------------- SSD: CB = C @ B^T per chunk (WMMA) --------------
__global__ __launch_bounds__(256)
void k_cb(const _Float16* __restrict__ C16, const _Float16* __restrict__ B16,
          float* __restrict__ CB)
{
  int wave = (int)((blockIdx.x * blockDim.x + threadIdx.x) >> 5);
  int lane = threadIdx.x & 31;
  if (wave >= NCK * 256) return;
  int c = wave >> 8, t = wave & 255;
  int ti = t >> 4, tj = t & 15;
  int lh = lane & 15, hg = lane >> 4;
  size_t base = (size_t)c * 256;
  const _Float16* Crow = C16 + (base + ti * 16 + lh) * DSTATE;
  const _Float16* Brow = B16 + (base + tj * 16 + lh) * DSTATE;
  v8f acc = {};
  for (int n0 = 0; n0 < DSTATE; n0 += 32) {
    v8h alo = *(const v8h*)(Crow + n0 + hg * 16);
    v8h ahi = *(const v8h*)(Crow + n0 + 8 + hg * 16);
    v16h a = joinh(alo, ahi);
    v16h b = *(const v16h*)(Brow + n0 + hg * 16);
    acc = wmma32(a, b, acc);
  }
#pragma unroll
  for (int r = 0; r < 8; r++)
    CB[(size_t)c * 65536 + (size_t)(ti * 16 + hg * 8 + r) * 256 + (tj * 16 + lh)] = acc[r];
}

// --------- SSD: Ydiag + Yoff, x-tile staged in LDS via async copy ----------
// grid: NCK*8*8 blocks; block = (c, h, ti-pair); 8 waves = 2 ti x 4 tp
__global__ __launch_bounds__(256)
void k_ssd_out(const float* __restrict__ CB, const _Float16* __restrict__ xhT16,
               const _Float16* __restrict__ C16, const _Float16* __restrict__ prev16,
               const float* __restrict__ dtpT, const float* __restrict__ dacsT,
               float* __restrict__ ybuf)
{
  __shared__ _Float16 xt[HEADDIM * 256];          // 32 KB: x tile [p][q]
  int blk = blockIdx.x;
  int sub = blk & 7;
  int h = (blk >> 3) & 7;
  int c = blk >> 6;
  int wave = threadIdx.x >> 5, lane = threadIdx.x & 31;
  int ti = sub * 2 + (wave >> 2);
  int tp = wave & 3;
  int lh = lane & 15, hg = lane >> 4;
  int ch = c * NHEADS + h;

  // async-stage the (c,h) x tile (64 x 256 halves) into LDS
  {
    const _Float16* src = xhT16 + (size_t)ch * (HEADDIM * 256);
    unsigned ldsbase = (unsigned)(uintptr_t)(&xt[0]);
    int t = threadIdx.x;
#pragma unroll
    for (int it = 0; it < 8; it++) {
      int e = t * 64 + it * 8;                    // halves
      unsigned lds = ldsbase + (unsigned)(e * 2);
      const void* g = (const void*)(src + e);
      asm volatile("global_load_async_to_lds_b128 %0, %1, off"
                   :: "v"(lds), "v"(g) : "memory");
    }
    asm volatile("s_wait_asynccnt 0x0" ::: "memory");
  }
  __syncthreads();

  int irow = ti * 16 + lh;                  // chunk-local row (A rows)
  int pcol = tp * 16 + lh;                  // output column inside head
  const float* dac = dacsT + (size_t)ch * 256;
  const float* dtr = dtpT + (size_t)ch * 256;
  float di = dac[irow];
  const float* CBrow = CB + (size_t)c * 65536 + (size_t)irow * 256;
  v8f acc = {};
  // part 1: Ydiag = M @ x,  M[i,j] = CB[i,j]*exp(dacs[i]-dacs[j])*dt[j], j<=i
  for (int j0 = 0; j0 < 256; j0 += 32) {
    v16h a, b;
#pragma unroll
    for (int i = 0; i < 8; i++) {
#pragma unroll
      for (int half = 0; half < 2; half++) {
        int jj = j0 + hg * 16 + i + half * 8;
        float mv = CBrow[jj] * __expf(di - dac[jj]) * dtr[jj];
        a[i + half * 8] = (_Float16)((jj <= irow) ? mv : 0.0f);
      }
    }
    b = *(const v16h*)(&xt[(size_t)pcol * 256 + j0 + hg * 16]);
    acc = wmma32(a, b, acc);
  }
  // part 2: Yoff = (C*exp(dacs)) @ prev^T
  float esc = __expf(di);
  const _Float16* Crow = C16 + ((size_t)c * 256 + irow) * DSTATE;
  const _Float16* Prow = prev16 + ((size_t)ch * HEADDIM + pcol) * DSTATE;
  for (int n0 = 0; n0 < DSTATE; n0 += 32) {
    v8h clo = *(const v8h*)(Crow + n0 + hg * 16);
    v8h chi = *(const v8h*)(Crow + n0 + 8 + hg * 16);
    v16h a, b;
#pragma unroll
    for (int i = 0; i < 8; i++) {
      a[i]     = (_Float16)((float)clo[i] * esc);
      a[i + 8] = (_Float16)((float)chi[i] * esc);
    }
    b = *(const v16h*)(Prow + n0 + hg * 16);
    acc = wmma32(a, b, acc);
  }
#pragma unroll
  for (int r = 0; r < 8; r++) {
    int l = c * 256 + ti * 16 + hg * 8 + r;
    if (l < LSEQ)
      ybuf[(size_t)l * DINNER + h * HEADDIM + pcol] = acc[r];
  }
}

// ------------------------- mamba: +D*x, silu(z) gate, RMS norm -------------
__global__ __launch_bounds__(256)
void k_gate_rms(float* __restrict__ ybuf, const _Float16* __restrict__ xhT16,
                const float* __restrict__ ZX, const float* __restrict__ Dp,
                const float* __restrict__ nw)
{
  int wid = (int)((blockIdx.x * blockDim.x + threadIdx.x) >> 5);
  int lane = threadIdx.x & 31;
  if (wid >= LSEQ) return;
  int c = wid >> 8, q = wid & 255;
  float vals[16];
  float ss = 0.0f;
#pragma unroll
  for (int t = 0; t < 16; t++) {
    int d = lane + t * 32;
    float xv = (float)xhT16[(((size_t)(c * NHEADS + (d >> 6))) * HEADDIM + (d & 63)) * 256 + q];
    float y = ybuf[(size_t)wid * DINNER + d] + Dp[d >> 6] * xv;
    float z = ZX[(size_t)wid * DINPROJ + d];
    float v = y * siluf(z);
    vals[t] = v;
    ss += v * v;
  }
  for (int o = 16; o > 0; o >>= 1) ss += __shfl_xor(ss, o);
  float r = rsqrtf(ss / (float)DINNER + 1e-5f);
#pragma unroll
  for (int t = 0; t < 16; t++) {
    int d = lane + t * 32;
    ybuf[(size_t)wid * DINNER + d] = vals[t] * r * nw[d];
  }
}

// ------------------------- concat [spec, sp2] ------------------------------
__global__ __launch_bounds__(256)
void k_concat(const float* __restrict__ A, const float* __restrict__ B,
              float* __restrict__ out)
{
  int idx = blockIdx.x * blockDim.x + threadIdx.x;
  if (idx >= LSEQ * 512) return;
  int l = idx >> 9, d = idx & 511;
  out[idx] = (d < 256) ? A[(size_t)l * 256 + d] : B[(size_t)l * 256 + (d - 256)];
}

// ---------------------------------------------------------------------------
// Workspace layout (float units; f16 buffers use halves/2 floats)
// ---------------------------------------------------------------------------
static constexpr size_t SL = LSEQ, SP = LPAD;
static constexpr size_t O_XN    = 0;                          // L*200 f32
static constexpr size_t O_XS    = O_XN    + SL * 200;
static constexpr size_t O_SPEC  = O_XS    + SL * 256;
static constexpr size_t O_SPC   = O_SPEC  + SL * 256;
static constexpr size_t O_SP2   = O_SPC   + SL * 256;
static constexpr size_t O_COMB  = O_SP2   + SL * 256;         // L*512
static constexpr size_t O_FU    = O_COMB  + SL * 512;
static constexpr size_t O_MOUT  = O_FU    + SL * 256;
static constexpr size_t O_T0    = O_MOUT  + SL * 256;
static constexpr size_t O_T1    = O_T0    + SL * 256;         // L*200
static constexpr size_t O_T2    = O_T1    + SL * 200;         // L*128
static constexpr size_t O_ZX    = O_T2    + SL * 128;         // L*1288
static constexpr size_t O_XBC   = O_ZX    + SL * 1288;        // L*768 (reused as ybuf)
static constexpr size_t O_DTP   = O_XBC   + SL * 768;         // Lp*8
static constexpr size_t O_DTPT  = O_DTP   + SP * 8;           // Lp*8 transposed
static constexpr size_t O_DACST = O_DTPT  + SP * 8;           // Lp*8 transposed
static constexpr size_t O_DEC   = O_DACST + SP * 8;           // NC*8
static constexpr size_t O_CBUF  = O_DEC   + (size_t)NCK * 8;  // NC*65536 f32
static constexpr size_t O_ST    = O_CBUF  + (size_t)NCK * 65536;
// f16 buffers (sizes in floats = halves/2)
static constexpr size_t O_PREV16 = O_ST     + (size_t)NCK * 65536;  // NC*65536 h
static constexpr size_t O_XHT    = O_PREV16 + (size_t)NCK * 32768;  // Lp*512 h
static constexpr size_t O_XWT    = O_XHT    + SP * 256;             // Lp*512 h
static constexpr size_t O_B16    = O_XWT    + SP * 256;             // Lp*128 h
static constexpr size_t O_C16    = O_B16    + SP * 64;              // Lp*128 h
static constexpr size_t O_BT16   = O_C16    + SP * 64;              // Lp*128 h
static constexpr size_t O_A16    = O_BT16   + SP * 64;              // MPAD*1824 h
static constexpr size_t O_WPK    = O_A16 + (size_t)MPAD * KCONVP / 2;

static void run_gemm16(const float* A, const float* W, const float* bias,
                       const float* scale, const float* shift, float smul,
                       float* C, int M, int N, int K, int act,
                       _Float16* A16, _Float16* Wp, hipStream_t s)
{
  int Kp = (K + 31) & ~31;
  int Mp = (M + 15) & ~15;
  int nkb = Kp >> 5;
  int ntile = (N + 15) >> 4;
  int totA = Mp * Kp;
  k_cvtA<<<(totA + 255) / 256, 256, 0, s>>>(A, A16, M, K, Kp, totA);
  int totW = ntile * nkb * 512;
  k_packW<<<(totW + 255) / 256, 256, 0, s>>>(W, Wp, K, N, nkb, totW);
  int tiles = ((M + 15) / 16) * ntile;
  k_gemm16<<<(tiles + 7) / 8, 256, 0, s>>>(A16, Wp, bias, scale, shift, smul,
                                           C, M, N, Kp, act);
}

static void run_mamba(const float* u, const float* in_w, const float* conv_w,
                      const float* conv_b, const float* dt_bias, const float* A_log,
                      const float* Dp, const float* norm_w, const float* out_w,
                      float* out, float* w, hipStream_t s)
{
  float*     zx     = w + O_ZX;
  float*     xbc    = w + O_XBC;                 // later reused as ybuf
  float*     dtp    = w + O_DTP;
  float*     dtpT   = w + O_DTPT;
  float*     dacsT  = w + O_DACST;
  float*     dec    = w + O_DEC;
  float*     CB     = w + O_CBUF;
  float*     st     = w + O_ST;
  _Float16*  prev16 = (_Float16*)(w + O_PREV16);
  _Float16*  xhT16  = (_Float16*)(w + O_XHT);
  _Float16*  xwT16  = (_Float16*)(w + O_XWT);
  _Float16*  B16    = (_Float16*)(w + O_B16);
  _Float16*  C16    = (_Float16*)(w + O_C16);
  _Float16*  BT16   = (_Float16*)(w + O_BT16);
  _Float16*  A16    = (_Float16*)(w + O_A16);
  _Float16*  Wp     = (_Float16*)(w + O_WPK);
  float*     ybuf   = xbc;

  run_gemm16(u, in_w, nullptr, nullptr, nullptr, 1.0f, zx,
             LSEQ, DINPROJ, DMODEL, 0, A16, Wp, s);
  k_conv1d<<<(LSEQ * CONVDIM + 255) / 256, 256, 0, s>>>(zx, conv_w, conv_b, xbc);
  k_pack<<<LPAD, 256, 0, s>>>(xbc, zx, dt_bias, B16, C16, dtp);
  k_xT<<<(LPAD * DINNER + 255) / 256, 256, 0, s>>>(xbc, xhT16);
  k_bT<<<(LPAD * DSTATE + 255) / 256, 256, 0, s>>>(xbc, BT16);
  k_cumsum<<<(NCK * NHEADS + 63) / 64, 64, 0, s>>>(dtp, A_log, dacsT, dtpT, dec);
  k_xwT<<<(LPAD * DINNER + 255) / 256, 256, 0, s>>>(xhT16, dtpT, dacsT, xwT16);
  k_states<<<NCK * NHEADS, 256, 0, s>>>(xwT16, BT16, st);
  k_recur<<<(NHEADS * HEADDIM * DSTATE + 255) / 256, 256, 0, s>>>(st, dec, prev16);
  k_cb<<<(NCK * 256 + 7) / 8, 256, 0, s>>>(C16, B16, CB);
  k_ssd_out<<<NCK * NHEADS * 8, 256, 0, s>>>(CB, xhT16, C16, prev16,
                                             dtpT, dacsT, ybuf);
  k_gate_rms<<<(LSEQ + 7) / 8, 256, 0, s>>>(ybuf, xhT16, zx, Dp, norm_w);
  run_gemm16(ybuf, out_w, nullptr, nullptr, nullptr, 1.0f, out,
             LSEQ, DMODEL, DINNER, 0, A16, Wp, s);
}

extern "C" void kernel_launch(void* const* d_in, const int* in_sizes, int n_in,
                              void* d_out, int out_size, void* d_ws, size_t ws_size,
                              hipStream_t stream)
{
  (void)in_sizes; (void)n_in; (void)out_size; (void)ws_size;
  const float* x        = (const float*)d_in[0];
  const float* ln0_w    = (const float*)d_in[1];
  const float* ln0_b    = (const float*)d_in[2];
  const float* sp1_w    = (const float*)d_in[3];
  const float* sp1_b    = (const float*)d_in[4];
  const float* sp1_g    = (const float*)d_in[5];
  const float* sp1_be   = (const float*)d_in[6];
  const float* m1_in_w  = (const float*)d_in[7];
  const float* m1_cw    = (const float*)d_in[8];
  const float* m1_cb    = (const float*)d_in[9];
  const float* m1_dtb   = (const float*)d_in[10];
  const float* m1_Alog  = (const float*)d_in[11];
  const float* m1_D     = (const float*)d_in[12];
  const float* m1_nw    = (const float*)d_in[13];
  const float* m1_ow    = (const float*)d_in[14];
  const float* m2_in_w  = (const float*)d_in[15];
  const float* m2_cw    = (const float*)d_in[16];
  const float* m2_cb    = (const float*)d_in[17];
  const float* m2_dtb   = (const float*)d_in[18];
  const float* m2_Alog  = (const float*)d_in[19];
  const float* m2_D     = (const float*)d_in[20];
  const float* m2_nw    = (const float*)d_in[21];
  const float* m2_ow    = (const float*)d_in[22];
  const float* ln1_w    = (const float*)d_in[23];
  const float* ln1_b    = (const float*)d_in[24];
  const float* sc3_w    = (const float*)d_in[25];
  const float* sc3_b    = (const float*)d_in[26];
  const float* sc3_g    = (const float*)d_in[27];
  const float* sc3_be   = (const float*)d_in[28];
  const float* sc1_w    = (const float*)d_in[29];
  const float* sc1_b    = (const float*)d_in[30];
  const float* sc1_g    = (const float*)d_in[31];
  const float* sc1_be   = (const float*)d_in[32];
  const float* fu_w     = (const float*)d_in[33];
  const float* fu_b     = (const float*)d_in[34];
  const float* fu_g     = (const float*)d_in[35];
  const float* fu_be    = (const float*)d_in[36];
  const float* ln2_w    = (const float*)d_in[37];
  const float* ln2_b    = (const float*)d_in[38];
  const float* cl1_w    = (const float*)d_in[39];
  const float* cl1_b    = (const float*)d_in[40];
  const float* cl2_w    = (const float*)d_in[41];
  const float* cl2_b    = (const float*)d_in[42];
  const float* cl3_w    = (const float*)d_in[43];
  const float* cl3_b    = (const float*)d_in[44];

  float* w = (float*)d_ws;
  float* out = (float*)d_out;
  _Float16* A16 = (_Float16*)(w + O_A16);
  _Float16* Wp  = (_Float16*)(w + O_WPK);
  const float BN_SCALE = 1.0f / sqrtf(1.0f + 1e-5f);

  k_layernorm<<<(LSEQ + 7) / 8, 256, 0, stream>>>(x, ln0_w, ln0_b, w + O_XN,
                                                  LSEQ, BANDS, 0);
  run_gemm16(w + O_XN, sp1_w, sp1_b, sp1_g, sp1_be, BN_SCALE, w + O_XS,
             LSEQ, DMODEL, BANDS, 1, A16, Wp, stream);
  run_mamba(w + O_XS, m1_in_w, m1_cw, m1_cb, m1_dtb, m1_Alog, m1_D, m1_nw, m1_ow,
            w + O_MOUT, w, stream);
  k_layernorm<<<(LSEQ + 7) / 8, 256, 0, stream>>>(w + O_MOUT, ln1_w, ln1_b,
                                                  w + O_SPEC, LSEQ, DMODEL, 1);
  {
    int tot = MPAD * KCONVP;
    k_im2col<<<(tot + 255) / 256, 256, 0, stream>>>(w + O_XN, A16);
    int nkb = KCONVP / 32, ntile = DMODEL / 16;
    int totW = ntile * nkb * 512;
    k_packW<<<(totW + 255) / 256, 256, 0, stream>>>(sc3_w, Wp, KCONV, DMODEL, nkb, totW);
    int tiles = ((LSEQ + 15) / 16) * ntile;
    k_gemm16<<<(tiles + 7) / 8, 256, 0, stream>>>(A16, Wp, sc3_b, sc3_g, sc3_be,
                                                  BN_SCALE, w + O_SPC,
                                                  LSEQ, DMODEL, KCONVP, 1);
  }
  run_gemm16(w + O_SPC, sc1_w, sc1_b, sc1_g, sc1_be, BN_SCALE, w + O_SP2,
             LSEQ, DMODEL, DMODEL, 1, A16, Wp, stream);
  k_concat<<<(LSEQ * 512 + 255) / 256, 256, 0, stream>>>(w + O_SPEC, w + O_SP2,
                                                         w + O_COMB);
  run_gemm16(w + O_COMB, fu_w, fu_b, fu_g, fu_be, BN_SCALE, w + O_FU,
             LSEQ, DMODEL, 2 * DMODEL, 1, A16, Wp, stream);
  run_mamba(w + O_FU, m2_in_w, m2_cw, m2_cb, m2_dtb, m2_Alog, m2_D, m2_nw, m2_ow,
            w + O_MOUT, w, stream);
  k_layernorm<<<(LSEQ + 7) / 8, 256, 0, stream>>>(w + O_MOUT, ln2_w, ln2_b,
                                                  w + O_T0, LSEQ, DMODEL, 1);
  run_gemm16(w + O_T0, cl1_w, cl1_b, nullptr, nullptr, 1.0f, w + O_T1,
             LSEQ, 200, DMODEL, 1, A16, Wp, stream);
  run_gemm16(w + O_T1, cl2_w, cl2_b, nullptr, nullptr, 1.0f, w + O_T2,
             LSEQ, 128, 200, 1, A16, Wp, stream);
  run_gemm16(w + O_T2, cl3_w, cl3_b, nullptr, nullptr, 1.0f, out,
             LSEQ, NCLS, 128, 0, A16, Wp, stream);
}
